// GRU_23587960389796
// MI455X (gfx1250) — matrix operand order repaired and checked by reading, not compiled, not run on hardware
//
#include <hip/hip_runtime.h>
#include <cstdint>
#include <cstddef>

#define D_BATCH 256
#define T_SEQ   512
#define I_DIM   128
#define H_DIM   512

typedef __attribute__((ext_vector_type(16))) __bf16        bf16x16;
typedef __attribute__((ext_vector_type(8)))  float         f32x8;
typedef __attribute__((ext_vector_type(4)))  unsigned int  u32x4;
typedef __attribute__((ext_vector_type(8)))  unsigned int  u32x8;

// ---------- helpers ----------

__device__ __forceinline__ unsigned short f2bf(float f) {
  unsigned int u = __float_as_uint(f);
  unsigned int r = u + 0x7FFFu + ((u >> 16) & 1u);  // round-to-nearest-even
  return (unsigned short)(r >> 16);
}

__device__ __forceinline__ bf16x16 ld2q(const unsigned short* p0, const unsigned short* p1) {
  u32x4 a = *(const u32x4*)p0;
  u32x4 b = *(const u32x4*)p1;
  union { u32x8 u; bf16x16 v; } f;
  f.u[0] = a[0]; f.u[1] = a[1]; f.u[2] = a[2]; f.u[3] = a[3];
  f.u[4] = b[0]; f.u[5] = b[1]; f.u[6] = b[2]; f.u[7] = b[3];
  return f.v;
}

// A fragment: 16 rows (M) x 32 K, bf16.  Lane l: row = m0 + (l&15);
// v16 elems 0..7  -> K = k0 + (l<16 ? 0 : 8) + 0..7
// v16 elems 8..15 -> K = k0 + (l<16 ? 16 : 24) + 0..7
__device__ __forceinline__ bf16x16 load_a(const unsigned short* base, int ld,
                                          int m0, int k0, int lane) {
  int r  = m0 + (lane & 15);
  int kk = k0 + ((lane >> 4) << 3);
  const unsigned short* p = base + (size_t)r * ld + kk;
  return ld2q(p, p + 16);
}

// B fragment: 32 K x 16 cols (N), B[k][n] = W[n][k] (W row-major [out,in]).
// Lane l: col = n0 + (l&15); elems 0..7 -> K = k0 + (l<16 ? 0 : 8) + 0..7, elems 8..15 -> K = k0 + (l<16 ? 16 : 24) + 0..7
__device__ __forceinline__ bf16x16 load_b(const unsigned short* base, int ld,
                                          int n0, int k0, int lane) {
  int n  = n0 + (lane & 15);
  int kk = k0 + ((lane >> 4) << 3);
  const unsigned short* p = base + (size_t)n * ld + kk;
  return ld2q(p, p + 16);
}

// Accumulate a 32x32 output tile: C += A[m0:m0+32, 0:K] * B(=W^T)[0:K, n0:n0+32]
__device__ __forceinline__ void mma32x32(const unsigned short* __restrict__ A, int lda,
                                         const unsigned short* __restrict__ B, int ldb,
                                         int m0, int n0, int K, int lane,
                                         f32x8& c00, f32x8& c01, f32x8& c10, f32x8& c11) {
  for (int k = 0; k < K; k += 32) {
    bf16x16 a0 = load_a(A, lda, m0,      k, lane);
    bf16x16 a1 = load_a(A, lda, m0 + 16, k, lane);
    bf16x16 b0 = load_b(B, ldb, n0,      k, lane);
    bf16x16 b1 = load_b(B, ldb, n0 + 16, k, lane);
    c00 = __builtin_amdgcn_wmma_f32_16x16x32_bf16(false, a0, false, b0, (short)0, c00, false, false);
    c01 = __builtin_amdgcn_wmma_f32_16x16x32_bf16(false, a0, false, b1, (short)0, c01, false, false);
    c10 = __builtin_amdgcn_wmma_f32_16x16x32_bf16(false, a1, false, b0, (short)0, c10, false, false);
    c11 = __builtin_amdgcn_wmma_f32_16x16x32_bf16(false, a1, false, b1, (short)0, c11, false, false);
  }
}

// ---------- setup kernels ----------

__global__ void cvt_f32_bf16(const float* __restrict__ src, unsigned short* __restrict__ dst, int n) {
  int i = blockIdx.x * blockDim.x + threadIdx.x;
  if (i < n) dst[i] = f2bf(src[i]);
}

// xbf[t][d][i] = bf16(x[d][t][i])
__global__ void x_tmajor_bf16(const float* __restrict__ x, unsigned short* __restrict__ xbf) {
  int gid = blockIdx.x * blockDim.x + threadIdx.x;   // T*D*I = 2^24
  int i = gid & (I_DIM - 1);
  int d = (gid >> 7) & (D_BATCH - 1);
  int t = gid >> 15;
  xbf[gid] = f2bf(x[((size_t)d * T_SEQ + t) * I_DIM + i]);
}

__global__ void zero_state(float* __restrict__ hfp, unsigned short* __restrict__ hbf) {
  int i = blockIdx.x * blockDim.x + threadIdx.x;
  if (i < D_BATCH * H_DIM) { hfp[i] = 0.0f; hbf[i] = 0; }
}

// ---------- scan step kernel A: r and z gates ----------
// grid: 32 blocks x 256 thr (8 waves). 256 wave-tiles = 2 gates x 8 Mtiles x 16 Ntiles.

__global__ __launch_bounds__(256) void gru_step_rz(
    const unsigned short* __restrict__ xbf,   // [T,D,I] bf16
    const unsigned short* __restrict__ hbf,   // [D,H]  bf16  (h_{t-1})
    const float*          __restrict__ hfp,   // [D,H]  f32   (h_{t-1})
    const unsigned short* __restrict__ Ur, const unsigned short* __restrict__ Wr,
    const float* __restrict__ Urb, const float* __restrict__ Wrb,
    const unsigned short* __restrict__ Uz, const unsigned short* __restrict__ Wz,
    const float* __restrict__ Uzb, const float* __restrict__ Wzb,
    unsigned short* __restrict__ rh_bf,       // [D,H] bf16  (r*h)
    float*          __restrict__ zbuf,        // [D,H] f32   (z)
    int t) {
  const int lane = threadIdx.x & 31;
  const int wave = threadIdx.x >> 5;
  const int idx  = blockIdx.x * 8 + wave;     // [0,256)
  const int gate = idx >> 7;                  // 0 = r, 1 = z
  const int rem  = idx & 127;
  const int m0   = (rem >> 4) * 32;
  const int n0   = (rem & 15) * 32;

  const unsigned short* U  = gate ? Uz  : Ur;
  const unsigned short* W  = gate ? Wz  : Wr;
  const float*          Ub = gate ? Uzb : Urb;
  const float*          Wb = gate ? Wzb : Wrb;

  f32x8 c00 = {}, c01 = {}, c10 = {}, c11 = {};
  const unsigned short* xt = xbf + (size_t)t * D_BATCH * I_DIM;
  mma32x32(xt,  I_DIM, U, I_DIM, m0, n0, I_DIM, lane, c00, c01, c10, c11);
  mma32x32(hbf, H_DIM, W, H_DIM, m0, n0, H_DIM, lane, c00, c01, c10, c11);

  const int rbase = m0 + (lane >> 4) * 8;
  const int cbase = n0 + (lane & 15);

  auto emit = [&](const f32x8& c, int mi, int ni) {
    const int col  = cbase + ni * 16;
    const float bias = Ub[col] + Wb[col];
    #pragma unroll
    for (int v = 0; v < 8; ++v) {
      const int row = rbase + mi * 16 + v;
      const float pre = c[v] + bias;
      const float s = 1.0f / (1.0f + expf(-pre));
      const size_t off = (size_t)row * H_DIM + col;
      if (gate == 0) rh_bf[off] = f2bf(s * hfp[off]);
      else           zbuf[off]  = s;
    }
  };
  emit(c00, 0, 0); emit(c01, 0, 1); emit(c10, 1, 0); emit(c11, 1, 1);
}

// ---------- scan step kernel B: h_bar + state update ----------
// grid: 16 blocks x 256 thr. 128 wave-tiles = 8 Mtiles x 16 Ntiles.

__global__ __launch_bounds__(256) void gru_step_h(
    const unsigned short* __restrict__ xbf,
    const unsigned short* __restrict__ rh_bf,  // [D,H] bf16
    const unsigned short* __restrict__ Uh, const unsigned short* __restrict__ Wh,
    const float* __restrict__ Uhb, const float* __restrict__ Whb,
    const float* __restrict__ zbuf,
    float*          __restrict__ hfp,          // in: h_{t-1}, out: h_t
    unsigned short* __restrict__ hbf,          // out: bf16(h_t)
    unsigned short* __restrict__ hs_bf,        // [T,D,H] bf16
    int t) {
  const int lane = threadIdx.x & 31;
  const int wave = threadIdx.x >> 5;
  const int idx  = blockIdx.x * 8 + wave;     // [0,128)
  const int m0   = (idx >> 4) * 32;
  const int n0   = (idx & 15) * 32;

  f32x8 c00 = {}, c01 = {}, c10 = {}, c11 = {};
  const unsigned short* xt = xbf + (size_t)t * D_BATCH * I_DIM;
  mma32x32(xt,    I_DIM, Uh, I_DIM, m0, n0, I_DIM, lane, c00, c01, c10, c11);
  mma32x32(rh_bf, H_DIM, Wh, H_DIM, m0, n0, H_DIM, lane, c00, c01, c10, c11);

  const int rbase = m0 + (lane >> 4) * 8;
  const int cbase = n0 + (lane & 15);
  unsigned short* hs_t = hs_bf + (size_t)t * D_BATCH * H_DIM;

  auto emit = [&](const f32x8& c, int mi, int ni) {
    const int col  = cbase + ni * 16;
    const float bias = Uhb[col] + Whb[col];
    #pragma unroll
    for (int v = 0; v < 8; ++v) {
      const int row = rbase + mi * 16 + v;
      const size_t off = (size_t)row * H_DIM + col;
      const float hb = tanhf(c[v] + bias);
      const float z  = zbuf[off];
      const float hn = (1.0f - z) * hfp[off] + z * hb;
      hfp[off] = hn;
      const unsigned short b = f2bf(hn);
      hbf[off]  = b;
      hs_t[off] = b;
    }
  };
  emit(c00, 0, 0); emit(c01, 0, 1); emit(c10, 1, 0); emit(c11, 1, 1);
}

// ---------- output projection + log_softmax ----------
// block = 256 thr (8 waves): 16 rows x 128 cols; wave w does cols [16w,16w+16).
// grid = (T*D)/16 = 8192 blocks.

__global__ __launch_bounds__(256) void out_logsoftmax(
    const unsigned short* __restrict__ hs_bf,  // [T*D, H] bf16
    const unsigned short* __restrict__ Vbf,    // [I, H]   bf16
    const float* __restrict__ Vb,              // [I]
    float* __restrict__ out) {                 // [D, T, I]
  __shared__ float smem[16][I_DIM];
  const int lane = threadIdx.x & 31;
  const int wave = threadIdx.x >> 5;
  const int row0 = blockIdx.x * 16;
  const int n0   = wave * 16;

  f32x8 c = {};
  for (int k = 0; k < H_DIM; k += 32) {
    bf16x16 a = load_a(hs_bf, H_DIM, row0, k, lane);
    bf16x16 b = load_b(Vbf,  H_DIM, n0,   k, lane);
    c = __builtin_amdgcn_wmma_f32_16x16x32_bf16(false, a, false, b, (short)0, c, false, false);
  }

  const int col  = n0 + (lane & 15);
  const float bias = Vb[col];
  #pragma unroll
  for (int v = 0; v < 8; ++v)
    smem[(lane >> 4) * 8 + v][col] = c[v] + bias;
  __syncthreads();

  // log-softmax over I=128 per row; wave w handles rows 2w, 2w+1.
  #pragma unroll
  for (int rr = 0; rr < 2; ++rr) {
    const int r = wave * 2 + rr;
    const float v0 = smem[r][lane];
    const float v1 = smem[r][lane + 32];
    const float v2 = smem[r][lane + 64];
    const float v3 = smem[r][lane + 96];
    float m = fmaxf(fmaxf(v0, v1), fmaxf(v2, v3));
    #pragma unroll
    for (int s = 16; s > 0; s >>= 1) m = fmaxf(m, __shfl_xor(m, s, 32));
    float e = expf(v0 - m) + expf(v1 - m) + expf(v2 - m) + expf(v3 - m);
    #pragma unroll
    for (int s = 16; s > 0; s >>= 1) e += __shfl_xor(e, s, 32);
    const float lz = m + logf(e);
    const int grow = row0 + r;
    const int t = grow >> 8;          // row = t*256 + d
    const int d = grow & 255;
    float* o = out + ((size_t)d * T_SEQ + t) * I_DIM;
    o[lane]      = v0 - lz;
    o[lane + 32] = v1 - lz;
    o[lane + 64] = v2 - lz;
    o[lane + 96] = v3 - lz;
  }
}

// ---------- host ----------

extern "C" void kernel_launch(void* const* d_in, const int* in_sizes, int n_in,
                              void* d_out, int out_size, void* d_ws, size_t ws_size,
                              hipStream_t stream) {
  const float* x    = (const float*)d_in[0];
  const float* Urw  = (const float*)d_in[1];
  const float* Urb  = (const float*)d_in[2];
  const float* Wrw  = (const float*)d_in[3];
  const float* Wrb  = (const float*)d_in[4];
  const float* Uzw  = (const float*)d_in[5];
  const float* Uzb  = (const float*)d_in[6];
  const float* Wzw  = (const float*)d_in[7];
  const float* Wzb  = (const float*)d_in[8];
  const float* Uhw  = (const float*)d_in[9];
  const float* Uhb  = (const float*)d_in[10];
  const float* Whw  = (const float*)d_in[11];
  const float* Whb  = (const float*)d_in[12];
  const float* Vw   = (const float*)d_in[13];
  const float* Vb   = (const float*)d_in[14];
  float* out = (float*)d_out;

  char* base = (char*)d_ws;
  size_t off = 0;
  auto carve = [&](size_t bytes) -> void* {
    void* p = base + off;
    off += (bytes + 255) & ~(size_t)255;
    return p;
  };

  unsigned short* xbf   = (unsigned short*)carve((size_t)T_SEQ * D_BATCH * I_DIM * 2);
  unsigned short* UrB   = (unsigned short*)carve((size_t)H_DIM * I_DIM * 2);
  unsigned short* UzB   = (unsigned short*)carve((size_t)H_DIM * I_DIM * 2);
  unsigned short* UhB   = (unsigned short*)carve((size_t)H_DIM * I_DIM * 2);
  unsigned short* WrB   = (unsigned short*)carve((size_t)H_DIM * H_DIM * 2);
  unsigned short* WzB   = (unsigned short*)carve((size_t)H_DIM * H_DIM * 2);
  unsigned short* WhB   = (unsigned short*)carve((size_t)H_DIM * H_DIM * 2);
  unsigned short* VB    = (unsigned short*)carve((size_t)I_DIM * H_DIM * 2);
  float*          hfp   = (float*)         carve((size_t)D_BATCH * H_DIM * 4);
  unsigned short* hbf   = (unsigned short*)carve((size_t)D_BATCH * H_DIM * 2);
  unsigned short* rhbf  = (unsigned short*)carve((size_t)D_BATCH * H_DIM * 2);
  float*          zbuf  = (float*)         carve((size_t)D_BATCH * H_DIM * 4);
  unsigned short* hsbf  = (unsigned short*)carve((size_t)T_SEQ * D_BATCH * H_DIM * 2);
  (void)ws_size; (void)n_in; (void)in_sizes; (void)out_size;

  // --- setup: convert to bf16, zero state ---
  x_tmajor_bf16<<<(T_SEQ * D_BATCH * I_DIM) / 256, 256, 0, stream>>>(x, xbf);
  cvt_f32_bf16<<<(H_DIM * I_DIM + 255) / 256, 256, 0, stream>>>(Urw, UrB, H_DIM * I_DIM);
  cvt_f32_bf16<<<(H_DIM * I_DIM + 255) / 256, 256, 0, stream>>>(Uzw, UzB, H_DIM * I_DIM);
  cvt_f32_bf16<<<(H_DIM * I_DIM + 255) / 256, 256, 0, stream>>>(Uhw, UhB, H_DIM * I_DIM);
  cvt_f32_bf16<<<(H_DIM * H_DIM + 255) / 256, 256, 0, stream>>>(Wrw, WrB, H_DIM * H_DIM);
  cvt_f32_bf16<<<(H_DIM * H_DIM + 255) / 256, 256, 0, stream>>>(Wzw, WzB, H_DIM * H_DIM);
  cvt_f32_bf16<<<(H_DIM * H_DIM + 255) / 256, 256, 0, stream>>>(Whw, WhB, H_DIM * H_DIM);
  cvt_f32_bf16<<<(I_DIM * H_DIM + 255) / 256, 256, 0, stream>>>(Vw,  VB,  I_DIM * H_DIM);
  zero_state<<<(D_BATCH * H_DIM) / 256, 256, 0, stream>>>(hfp, hbf);

  // --- recurrent scan ---
  for (int t = 0; t < T_SEQ; ++t) {
    gru_step_rz<<<32, 256, 0, stream>>>(xbf, hbf, hfp,
                                        UrB, WrB, Urb, Wrb,
                                        UzB, WzB, Uzb, Wzb,
                                        rhbf, zbuf, t);
    gru_step_h<<<16, 256, 0, stream>>>(xbf, rhbf, UhB, WhB, Uhb, Whb,
                                       zbuf, hfp, hbf, hsbf, t);
  }

  // --- output projection + log_softmax, writes [D,T,I] directly ---
  out_logsoftmax<<<(T_SEQ * D_BATCH) / 16, 256, 0, stream>>>(hsbf, VB, Vb, out);
}
